// VQEmbeddingEMA_14010183319980
// MI455X (gfx1250) — compile-verified, matrix operand order checked by reading, not run on hardware
//
#include <hip/hip_runtime.h>
#include <hip/hip_bf16.h>
#include <stddef.h>

#define K_CODES 2048
#define DIM 512
#define N_VECS 65536
#define DECAYF 0.99f
#define EPSF 1e-5f
#define RG 4  // row-groups (of 16 rows) per block

typedef __attribute__((ext_vector_type(16))) __bf16 v16bf;
typedef __attribute__((ext_vector_type(8)))  __bf16 v8bf;
typedef __attribute__((ext_vector_type(8)))  float  v8f;

// ---------------- Kernel 1: split embedding into bf16 hi/lo + row norms ----
__global__ __launch_bounds__(256) void conv_embed_kernel(
    const float* __restrict__ E, __bf16* __restrict__ Ehi,
    __bf16* __restrict__ Elo, float* __restrict__ enorm) {
  __shared__ float red[256];
  const int k = blockIdx.x;
  float s = 0.f;
  for (int d = threadIdx.x; d < DIM; d += 256) {
    float x = E[(size_t)k * DIM + d];
    __bf16 h = (__bf16)x;
    __bf16 l = (__bf16)(x - (float)h);
    Ehi[(size_t)k * DIM + d] = h;
    Elo[(size_t)k * DIM + d] = l;
    s += x * x;
  }
  red[threadIdx.x] = s;
  __syncthreads();
  for (int o = 128; o > 0; o >>= 1) {
    if (threadIdx.x < o) red[threadIdx.x] += red[threadIdx.x + o];
    __syncthreads();
  }
  if (threadIdx.x == 0) enorm[k] = red[0];
}

// ---------------- Kernel 2: init EMA outputs + zero accumulators -----------
__global__ __launch_bounds__(256) void init_out_kernel(
    const float* __restrict__ cluster_size, const float* __restrict__ embed_avg,
    float* __restrict__ cs_out, float* __restrict__ avg_out,
    float* __restrict__ loss_acc) {
  size_t i = (size_t)blockIdx.x * 256 + threadIdx.x;  // over K*D
  avg_out[i] = DECAYF * embed_avg[i];
  if (i < K_CODES) cs_out[i] = DECAYF * cluster_size[i];
  if (i == 0) loss_acc[0] = 0.f;
}

// ---------------- Kernel 3: WMMA distances + argmin ------------------------
// Block = 8 waves, owns RG*16 = 64 rows of z_e (hi/lo bf16 in LDS).
// Wave w covers codes [w*256, (w+1)*256) in 16-code tiles; each B tile is
// reused against all RG row-groups (12 WMMAs per 64B of B traffic).
__global__ __launch_bounds__(256) void vq_dist_kernel(
    const float* __restrict__ z_e, const __bf16* __restrict__ Ehi,
    const __bf16* __restrict__ Elo, const float* __restrict__ enorm,
    int* __restrict__ idx_int, float* __restrict__ idx_f) {
  __shared__ __bf16 Ahi[RG * 16 * DIM];  // 64 KB
  __shared__ __bf16 Alo[RG * 16 * DIM];  // 64 KB
  __shared__ float wbv[8][RG * 16];
  __shared__ int wbi[8][RG * 16];

  const int tid = threadIdx.x;
  const int row0 = blockIdx.x * (RG * 16);

  // Stage this block's 64 z_e rows in LDS as bf16 hi/lo.
  for (int i = tid; i < RG * 16 * DIM; i += 256) {
    float x = z_e[(size_t)row0 * DIM + i];
    __bf16 h = (__bf16)x;
    Ahi[i] = h;
    Alo[i] = (__bf16)(x - (float)h);
  }
  __syncthreads();

  const int wave = tid >> 5;
  const int lane = tid & 31;
  const int col  = lane & 15;   // N for B/C operands, M row for A operand
  const int half = lane >> 4;
  const int koff = half * 8;    // per-ISA 16-bit operand layout

  float bestv[RG][8];
  int   besti[RG][8];
#pragma unroll
  for (int rg = 0; rg < RG; ++rg)
#pragma unroll
    for (int i = 0; i < 8; ++i) { bestv[rg][i] = 3.4e38f; besti[rg][i] = 0; }

  for (int j = 0; j < 16; ++j) {
    const int code = (wave * 16 + j) * 16 + col;
    const float en = enorm[code];
    v8f acc[RG];
#pragma unroll
    for (int rg = 0; rg < RG; ++rg) acc[rg] = (v8f){};

    for (int dc = 0; dc < 16; ++dc) {
      const int d0 = dc * 32;
      const size_t boff = (size_t)code * DIM + d0 + koff;
      v8bf b0 = *(const v8bf*)&Ehi[boff];
      v8bf b1 = *(const v8bf*)&Ehi[boff + 16];
      v8bf b2 = *(const v8bf*)&Elo[boff];
      v8bf b3 = *(const v8bf*)&Elo[boff + 16];
      v16bf bh, bl;
#pragma unroll
      for (int q = 0; q < 8; ++q) {
        bh[q] = b0[q]; bh[q + 8] = b1[q];
        bl[q] = b2[q]; bl[q + 8] = b3[q];
      }
#pragma unroll
      for (int rg = 0; rg < RG; ++rg) {
        const int aoff = (rg * 16 + col) * DIM + d0 + koff;
        v8bf a0 = *(const v8bf*)&Ahi[aoff];
        v8bf a1 = *(const v8bf*)&Ahi[aoff + 16];
        v8bf a2 = *(const v8bf*)&Alo[aoff];
        v8bf a3 = *(const v8bf*)&Alo[aoff + 16];
        v16bf ah, al;
#pragma unroll
        for (int q = 0; q < 8; ++q) {
          ah[q] = a0[q]; ah[q + 8] = a1[q];
          al[q] = a2[q]; al[q + 8] = a3[q];
        }
        acc[rg] = __builtin_amdgcn_wmma_f32_16x16x32_bf16(
            false, ah, false, bh, (short)0, acc[rg], false, false);
        acc[rg] = __builtin_amdgcn_wmma_f32_16x16x32_bf16(
            false, ah, false, bl, (short)0, acc[rg], false, false);
        acc[rg] = __builtin_amdgcn_wmma_f32_16x16x32_bf16(
            false, al, false, bh, (short)0, acc[rg], false, false);
      }
    }
#pragma unroll
    for (int rg = 0; rg < RG; ++rg) {
#pragma unroll
      for (int i = 0; i < 8; ++i) {
        float d = en - 2.0f * acc[rg][i];
        if (d < bestv[rg][i] || (d == bestv[rg][i] && code < besti[rg][i])) {
          bestv[rg][i] = d;
          besti[rg][i] = code;
        }
      }
    }
  }

  // Reduce across the 16 lanes of each half (same row, different codes).
#pragma unroll
  for (int rg = 0; rg < RG; ++rg) {
#pragma unroll
    for (int i = 0; i < 8; ++i) {
      float v = bestv[rg][i];
      int ix = besti[rg][i];
#pragma unroll
      for (int off = 8; off >= 1; off >>= 1) {
        float ov = __shfl_xor(v, off, 32);
        int oi = __shfl_xor(ix, off, 32);
        if (ov < v || (ov == v && oi < ix)) { v = ov; ix = oi; }
      }
      if (col == 0) {
        int m = rg * 16 + i + 8 * half;  // C-layout: VGPR i -> row i / 8+i
        wbv[wave][m] = v;
        wbi[wave][m] = ix;
      }
    }
  }
  __syncthreads();
  if (tid < RG * 16) {
    float v = wbv[0][tid];
    int ix = wbi[0][tid];
#pragma unroll
    for (int w = 1; w < 8; ++w) {
      float ov = wbv[w][tid];
      int oi = wbi[w][tid];
      if (ov < v || (ov == v && oi < ix)) { v = ov; ix = oi; }
    }
    idx_int[row0 + tid] = ix;
    idx_f[row0 + tid] = (float)ix;
  }
}

// ---------------- Kernel 4: gather z_q + loss ------------------------------
__global__ __launch_bounds__(256) void gather_loss_kernel(
    const float* __restrict__ z_e, const float* __restrict__ E,
    const int* __restrict__ idx_int, float* __restrict__ zq_out,
    float* __restrict__ loss_acc) {
  __shared__ float red[256];
  const int n = blockIdx.x;
  const int c = idx_int[n];
  float s = 0.f;
  for (int d = threadIdx.x; d < DIM; d += 256) {
    float e = E[(size_t)c * DIM + d];
    float x = z_e[(size_t)n * DIM + d];
    zq_out[(size_t)n * DIM + d] = e;
    float df = x - e;
    s += df * df;
  }
  red[threadIdx.x] = s;
  __syncthreads();
  for (int o = 128; o > 0; o >>= 1) {
    if (threadIdx.x < o) red[threadIdx.x] += red[threadIdx.x + o];
    __syncthreads();
  }
  if (threadIdx.x == 0) atomicAdd(loss_acc, red[0]);
}

// ---------------- Kernel 5: segment-sum scatter ----------------------------
__global__ __launch_bounds__(256) void scatter_kernel(
    const float* __restrict__ z_e, const int* __restrict__ idx_int,
    float* __restrict__ avg_out, float* __restrict__ cs_out) {
  size_t i = (size_t)blockIdx.x * 256 + threadIdx.x;  // over N*D
  int n = (int)(i / DIM);
  int d = (int)(i % DIM);
  int c = idx_int[n];
  atomicAdd(&avg_out[(size_t)c * DIM + d], (1.0f - DECAYF) * z_e[i]);
  if (d == 0) atomicAdd(&cs_out[c], 1.0f - DECAYF);
}

// ---------------- Kernel 6: n = sum(cs), finalize loss ---------------------
__global__ __launch_bounds__(256) void reduce_n_kernel(
    const float* __restrict__ cs_out, float* __restrict__ nsum,
    const float* __restrict__ loss_acc, float* __restrict__ loss_out) {
  __shared__ float red[256];
  float s = 0.f;
  for (int k = threadIdx.x; k < K_CODES; k += 256) s += cs_out[k];
  red[threadIdx.x] = s;
  __syncthreads();
  for (int o = 128; o > 0; o >>= 1) {
    if (threadIdx.x < o) red[threadIdx.x] += red[threadIdx.x + o];
    __syncthreads();
  }
  if (threadIdx.x == 0) {
    nsum[0] = red[0];
    loss_out[0] = 1.25f * loss_acc[0] / ((float)N_VECS * (float)DIM);
  }
}

// ---------------- Kernel 7: normalized embedding ---------------------------
__global__ __launch_bounds__(256) void finalize_embed_kernel(
    const float* __restrict__ avg_out, const float* __restrict__ cs_out,
    const float* __restrict__ nsum, float* __restrict__ emb_out) {
  size_t i = (size_t)blockIdx.x * 256 + threadIdx.x;  // over K*D
  int k = (int)(i / DIM);
  float n = nsum[0];
  float denom = (cs_out[k] + EPSF) / (n + (float)K_CODES * EPSF) * n;
  emb_out[i] = avg_out[i] / denom;
}

extern "C" void kernel_launch(void* const* d_in, const int* in_sizes, int n_in,
                              void* d_out, int out_size, void* d_ws,
                              size_t ws_size, hipStream_t stream) {
  const float* z_e          = (const float*)d_in[0];
  const float* embedding    = (const float*)d_in[1];
  const float* cluster_size = (const float*)d_in[2];
  const float* embed_avg    = (const float*)d_in[3];

  float* out = (float*)d_out;
  float* zq_out   = out;                               // 16*4096*512
  float* idx_out  = zq_out + (size_t)N_VECS * DIM;     // 65536
  float* loss_out = idx_out + N_VECS;                  // 1
  float* emb_out  = loss_out + 1;                      // K*D
  float* cs_out   = emb_out + (size_t)K_CODES * DIM;   // K
  float* avg_out  = cs_out + K_CODES;                  // K*D

  char* ws = (char*)d_ws;
  __bf16* Ehi   = (__bf16*)ws;                                    // 2 MB
  __bf16* Elo   = (__bf16*)(ws + (size_t)K_CODES * DIM * 2);      // 2 MB
  float* enorm  = (float*)(ws + (size_t)K_CODES * DIM * 4);       // 8 KB
  int* idx_int  = (int*)((char*)enorm + K_CODES * 4);             // 256 KB
  float* loss_acc = (float*)((char*)idx_int + (size_t)N_VECS * 4);
  float* nsum     = loss_acc + 1;

  conv_embed_kernel<<<K_CODES, 256, 0, stream>>>(embedding, Ehi, Elo, enorm);
  init_out_kernel<<<(K_CODES * DIM) / 256, 256, 0, stream>>>(
      cluster_size, embed_avg, cs_out, avg_out, loss_acc);
  vq_dist_kernel<<<N_VECS / (RG * 16), 256, 0, stream>>>(z_e, Ehi, Elo, enorm,
                                                         idx_int, idx_out);
  gather_loss_kernel<<<N_VECS, 256, 0, stream>>>(z_e, embedding, idx_int,
                                                 zq_out, loss_acc);
  scatter_kernel<<<(int)(((size_t)N_VECS * DIM) / 256), 256, 0, stream>>>(
      z_e, idx_int, avg_out, cs_out);
  reduce_n_kernel<<<1, 256, 0, stream>>>(cs_out, nsum, loss_acc, loss_out);
  finalize_embed_kernel<<<(K_CODES * DIM) / 256, 256, 0, stream>>>(
      avg_out, cs_out, nsum, emb_out);
}
